// TATT_30064771072786
// MI455X (gfx1250) — matrix operand order repaired and checked by reading, not compiled, not run on hardware
//
#include <hip/hip_runtime.h>
#include <math.h>

// ---------------- CDNA5 WMMA types ----------------
typedef __bf16 bf16_t;
typedef bf16_t v16bf __attribute__((ext_vector_type(16)));
typedef float  v8f   __attribute__((ext_vector_type(8)));
typedef unsigned int v4u __attribute__((ext_vector_type(4)));

#define DEV static __device__ __forceinline__

// ---------------- problem constants ----------------
constexpr int BATCH = 16;
constexpr int CXH   = 256;                 // channels of x (== channels of tem)
constexpr int CIN   = 512;                 // concat channels
constexpr int COUT  = 512;
constexpr int NN    = 170;
constexpr int TT    = 64;
constexpr int TOK   = NN * TT;             // 10880 tokens per batch
constexpr int HEADS = 8;
constexpr int HD    = 64;                  // head dim
constexpr int PROB  = TT * HD;             // 4096 elems per (b,head,n) tile
constexpr float SMALLV = -32767.0f;

// LDS row strides (ushorts): 16B-multiple and bank-conflict-free
constexpr int GPAD = 56;   // GEMM tiles: 28-dword stride -> 28*ln mod 64 all distinct
constexpr int APAD = 72;   // attn tiles: 36-dword stride -> 36*ln mod 64 all distinct

// fp32 -> bf16 round-to-nearest-even
DEV unsigned short f2bf(float f) {
    unsigned int u = __builtin_bit_cast(unsigned int, f);
    u += 0x7FFFu + ((u >> 16) & 1u);
    return (unsigned short)(u >> 16);
}
DEV unsigned int pack2bf(float a, float b) {
    return (unsigned int)f2bf(a) | ((unsigned int)f2bf(b) << 16);
}

struct FragBits { v4u lo, hi; };

// A-matrix fragment (16x32): LDS tile row-major [M][K]; row = &tile[m][0]
DEV v16bf ldfragA(const unsigned short* row, int ks, int h) {
    FragBits f;
    f.lo = *(const v4u*)(row + ks + 8 * h);
    f.hi = *(const v4u*)(row + ks + 16 + 8 * h);
    return __builtin_bit_cast(v16bf, f);
}
// B-matrix fragment (32x16): LDS tile transposed [N][K]; row = &tile[n][0]
DEV v16bf ldfragB(const unsigned short* row, int ks, int h) {
    FragBits f;
    f.lo = *(const v4u*)(row + ks + 8 * h);
    f.hi = *(const v4u*)(row + ks + 16 + 8 * h);
    return __builtin_bit_cast(v16bf, f);
}

// =====================================================================
// Kernel 1: FUSED QKV projection.  One block stages the X tile once and
// computes q,k,v 64x64 tiles (12 WMMA per K-step, 96 accum VGPRs).
// Layouts: q,k -> [b][head][n][t][d]   (attention A / B^T friendly)
//          v   -> [b][head][n][d][t]   (attention B friendly)
// grid: (NN, HEADS, BATCH)  [m-tile == head since 64 outs per head]
// block: 128 (4 waves)
// =====================================================================
__global__ __launch_bounds__(128)
void k_qkv(const float* __restrict__ x, const float* __restrict__ tem,
           const float* __restrict__ Wq, const float* __restrict__ bq,
           const float* __restrict__ Wk, const float* __restrict__ bk,
           const float* __restrict__ Wv, const float* __restrict__ bv,
           unsigned short* __restrict__ ws_qkv)
{
    const int tid  = threadIdx.x;
    const int wave = tid >> 5;          // wave32
    const int lane = tid & 31;
    const int h    = lane >> 4;
    const int ln   = lane & 15;

    const int n    = blockIdx.x;
    const int head = blockIdx.y;
    const int m0   = head * 64;
    const int b    = blockIdx.z;
    const int tok0 = n * TT;

    const float* Ws[3]     = {Wq, Wk, Wv};
    const float* biases[3] = {bq, bk, bv};

    __shared__ __align__(16) unsigned short Wl[3][64][GPAD]; // [o][c] (A)
    __shared__ __align__(16) unsigned short Xl[64][GPAD];    // [tok][c] (B^T)

    v8f acc[3][4];
    #pragma unroll
    for (int w3 = 0; w3 < 3; ++w3)
        #pragma unroll
        for (int j = 0; j < 4; ++j)
            acc[w3][j] = (v8f){0.f, 0.f, 0.f, 0.f, 0.f, 0.f, 0.f, 0.f};

    for (int kt = 0; kt < CIN / 32; ++kt) {
        const int k0 = kt * 32;

        if (kt + 1 < CIN / 32) {           // prefetch next streamed X tile
            int chp = k0 + 32 + (tid & 31);
            const float* srcp = (chp < CXH) ? x : tem;
            __builtin_prefetch(&srcp[((size_t)b * CXH + (chp & (CXH - 1))) * TOK + tok0], 0, 0);
        }

        // stage 3 W tiles 64x32 (f32 -> bf16, dword-packed writes)
        #pragma unroll
        for (int w3 = 0; w3 < 3; ++w3) {
            #pragma unroll
            for (int i = 0; i < 8; ++i) {
                int idx = i * 128 + tid;            // 0..1023 dword slots
                int r = idx >> 4, c2 = idx & 15;
                float2 wv = *(const float2*)&Ws[w3][(size_t)(m0 + r) * CIN + k0 + 2 * c2];
                *(unsigned int*)&Wl[w3][r][2 * c2] = pack2bf(wv.x, wv.y);
            }
        }
        // stage X tile transposed [tok][c]: channel pair per thread
        #pragma unroll
        for (int i = 0; i < 8; ++i) {
            int idx = i * 128 + tid;
            int c2 = idx >> 6, t = idx & 63;
            int ch = k0 + 2 * c2;
            const float* src = (ch < CXH) ? x : tem;
            size_t g = ((size_t)b * CXH + (ch & (CXH - 1))) * TOK + tok0 + t;
            *(unsigned int*)&Xl[t][2 * c2] = pack2bf(src[g], src[g + TOK]);
        }
        __syncthreads();

        v16bf a0 = ldfragA(&Wl[0][wave * 16 + ln][0], 0, h);
        v16bf a1 = ldfragA(&Wl[1][wave * 16 + ln][0], 0, h);
        v16bf a2 = ldfragA(&Wl[2][wave * 16 + ln][0], 0, h);
        #pragma unroll
        for (int j = 0; j < 4; ++j) {
            v16bf bm = ldfragB(&Xl[j * 16 + ln][0], 0, h);
            acc[0][j] = __builtin_amdgcn_wmma_f32_16x16x32_bf16(
                false, a0, false, bm, (short)0, acc[0][j], false, false);
            acc[1][j] = __builtin_amdgcn_wmma_f32_16x16x32_bf16(
                false, a1, false, bm, (short)0, acc[1][j], false, false);
            acc[2][j] = __builtin_amdgcn_wmma_f32_16x16x32_bf16(
                false, a2, false, bm, (short)0, acc[2][j], false, false);
        }
        __syncthreads();
    }

    // ---- epilogue ----
    const size_t QSZ  = (size_t)BATCH * COUT * TOK;   // elems per tensor
    const size_t prob = (((size_t)b * HEADS + head) * NN + n) * PROB;
    const int d0 = wave * 16 + 8 * h;                 // C layout: d = d0 + r

    // q, k: [t][d] layout -> one b128 per tile (8 d-consecutive values)
    #pragma unroll
    for (int w3 = 0; w3 < 2; ++w3) {
        float bvv[8];
        #pragma unroll
        for (int r = 0; r < 8; ++r) bvv[r] = biases[w3][m0 + d0 + r];
        unsigned short* dst = ws_qkv + (size_t)w3 * QSZ + prob;
        #pragma unroll
        for (int j = 0; j < 4; ++j) {
            int t = j * 16 + ln;
            v4u pk;
            #pragma unroll
            for (int jj = 0; jj < 4; ++jj)
                pk[jj] = pack2bf(acc[w3][j][2 * jj]     + bvv[2 * jj],
                                 acc[w3][j][2 * jj + 1] + bvv[2 * jj + 1]);
            *(v4u*)(dst + (size_t)t * HD + d0) = pk;
        }
    }
    // v: [d][t] layout -> scalar b16 stores (t contiguous per half-row only)
    {
        float bvv[8];
        #pragma unroll
        for (int r = 0; r < 8; ++r) bvv[r] = bv[m0 + d0 + r];
        unsigned short* dst = ws_qkv + 2 * QSZ + prob;
        #pragma unroll
        for (int j = 0; j < 4; ++j) {
            int t = j * 16 + ln;
            #pragma unroll
            for (int r = 0; r < 8; ++r)
                dst[(size_t)(d0 + r) * TT + t] = f2bf(acc[2][j][r] + bvv[r]);
        }
    }
}

// =====================================================================
// Kernel 2: causal attention per (head, b, n).  T = d = 64.
// All staging is now straight b128 copies (layouts match fragments).
// grid: (NN, BATCH, HEADS), block: 256 (8 waves)
// =====================================================================
__global__ __launch_bounds__(256)
void k_attn(const unsigned short* __restrict__ ws_qkv,
            unsigned short* __restrict__ ctx)
{
    const int tid  = threadIdx.x;
    const int wave = tid >> 5;
    const int lane = tid & 31;
    const int h    = lane >> 4;
    const int ln   = lane & 15;

    const int n    = blockIdx.x;
    const int b    = blockIdx.y;
    const int head = blockIdx.z;

    __shared__ __align__(16) unsigned short Ql[64][APAD];  // [t][d] (A)
    __shared__ __align__(16) unsigned short Kl[64][APAD];  // [s][d] (B^T)
    __shared__ __align__(16) unsigned short Vl[64][APAD];  // [d][s] (B^T)
    __shared__ __align__(16) unsigned short At[64][APAD];  // [t][s] (A)
    __shared__ float Sc[64][65];                           // scores [t][s]

    const size_t QSZ  = (size_t)BATCH * COUT * TOK;
    const size_t prob = (((size_t)b * HEADS + head) * NN + n) * PROB;
    const v4u* q4 = (const v4u*)(ws_qkv + prob);             // [t][d] 8KB block
    const v4u* k4 = (const v4u*)(ws_qkv + QSZ + prob);       // [s][d]
    const v4u* v4 = (const v4u*)(ws_qkv + 2 * QSZ + prob);   // [d][s]

    // stage: pure b128 copies, 2 per tensor per thread (512 v4u each)
    #pragma unroll
    for (int i = 0; i < 2; ++i) {
        int idx = i * 256 + tid;                 // 0..511
        int r = idx >> 3, col = (idx & 7) * 8;   // 8 ushorts per v4u
        *(v4u*)&Ql[r][col] = q4[idx];
        *(v4u*)&Kl[r][col] = k4[idx];
        *(v4u*)&Vl[r][col] = v4[idx];
    }
    __syncthreads();

    // ---- scores = q . k^T / 8 : 16 tiles, 2 per wave ----
    #pragma unroll
    for (int it = 0; it < 2; ++it) {
        int tl = wave * 2 + it;
        int ti = tl >> 2, si = tl & 3;
        v8f acc = (v8f){0.f, 0.f, 0.f, 0.f, 0.f, 0.f, 0.f, 0.f};
        #pragma unroll
        for (int ks = 0; ks < 64; ks += 32) {
            v16bf a  = ldfragA(&Ql[ti * 16 + ln][0], ks, h);
            v16bf bm = ldfragB(&Kl[si * 16 + ln][0], ks, h);
            acc = __builtin_amdgcn_wmma_f32_16x16x32_bf16(
                false, a, false, bm, (short)0, acc, false, false);
        }
        #pragma unroll
        for (int r = 0; r < 8; ++r)
            Sc[ti * 16 + 8 * h + r][si * 16 + ln] = acc[r] * 0.125f; // 1/sqrt(64)
    }
    __syncthreads();

    // ---- causal softmax, one thread per row ----
    if (tid < 64) {
        int t = tid;
        float mx = SMALLV;
        for (int s = 0; s <= t; ++s) mx = fmaxf(mx, Sc[t][s]);
        float sum = 0.f;
        for (int s = 0; s <= t; ++s) sum += __expf(Sc[t][s] - mx);
        float inv = 1.0f / sum;
        for (int s = 0; s < 32; ++s) {
            float p0 = (2 * s     <= t) ? __expf(Sc[t][2 * s]     - mx) * inv : 0.f;
            float p1 = (2 * s + 1 <= t) ? __expf(Sc[t][2 * s + 1] - mx) * inv : 0.f;
            *(unsigned int*)&At[t][2 * s] = pack2bf(p0, p1);
        }
    }
    __syncthreads();

    // ---- out = attn . v : 16 tiles, 2 per wave ----
    #pragma unroll
    for (int it = 0; it < 2; ++it) {
        int tl = wave * 2 + it;
        int ti = tl >> 2, di = tl & 3;
        v8f acc = (v8f){0.f, 0.f, 0.f, 0.f, 0.f, 0.f, 0.f, 0.f};
        #pragma unroll
        for (int ks = 0; ks < 64; ks += 32) {
            v16bf a  = ldfragA(&At[ti * 16 + ln][0], ks, h);
            v16bf bm = ldfragB(&Vl[di * 16 + ln][0], ks, h);
            acc = __builtin_amdgcn_wmma_f32_16x16x32_bf16(
                false, a, false, bm, (short)0, acc, false, false);
        }
        // ctx layout [d][t]: 8 consecutive t per lane -> one b128
        int d  = di * 16 + ln;
        int t0 = ti * 16 + 8 * h;
        v4u pk;
        #pragma unroll
        for (int jj = 0; jj < 4; ++jj)
            pk[jj] = pack2bf(acc[2 * jj], acc[2 * jj + 1]);
        *(v4u*)(ctx + prob + (size_t)d * TT + t0) = pk;
    }
}

// =====================================================================
// Kernel 3: output projection + bias + ReLU.  y fp32 [b][o][tok]
// ctx read from compact per-problem [d][t] blocks.
// grid: (NN, COUT/64, BATCH), block: 128 (4 waves)
// =====================================================================
__global__ __launch_bounds__(128)
void k_proj(const unsigned short* __restrict__ ctx,
            const float* __restrict__ Wo, const float* __restrict__ bo,
            float* __restrict__ y)
{
    const int tid  = threadIdx.x;
    const int wave = tid >> 5;
    const int lane = tid & 31;
    const int h    = lane >> 4;
    const int ln   = lane & 15;

    const int n    = blockIdx.x;
    const int tok0 = n * TT;
    const int m0   = blockIdx.y * 64;
    const int b    = blockIdx.z;

    __shared__ __align__(16) unsigned short Wl[64][GPAD];  // [o][c]
    __shared__ __align__(16) unsigned short Xl[64][GPAD];  // [tok][c]

    v8f acc[4];
    #pragma unroll
    for (int j = 0; j < 4; ++j)
        acc[j] = (v8f){0.f, 0.f, 0.f, 0.f, 0.f, 0.f, 0.f, 0.f};

    for (int kt = 0; kt < COUT / 32; ++kt) {
        const int k0 = kt * 32;

        if (kt + 1 < COUT / 32)
            __builtin_prefetch(&Wo[(size_t)(m0 + (tid & 63)) * COUT + k0 + 32], 0, 0);

        // stage W tile (f32 -> bf16, dword-packed)
        #pragma unroll
        for (int i = 0; i < 8; ++i) {
            int idx = i * 128 + tid;
            int r = idx >> 4, c2 = idx & 15;
            float2 wv = *(const float2*)&Wo[(size_t)(m0 + r) * COUT + k0 + 2 * c2];
            *(unsigned int*)&Wl[r][2 * c2] = pack2bf(wv.x, wv.y);
        }
        // stage X tile transposed [tok][c] from ctx [b][head][n][d][t]
        #pragma unroll
        for (int i = 0; i < 8; ++i) {
            int idx = i * 128 + tid;
            int c2 = idx >> 6, t = idx & 63;
            int ch   = k0 + 2 * c2;
            int head = ch >> 6, d0 = ch & 63;
            size_t g = ((((size_t)b * HEADS + head) * NN + n) * HD + d0) * TT + t;
            unsigned int w = (unsigned int)ctx[g] | ((unsigned int)ctx[g + TT] << 16);
            *(unsigned int*)&Xl[t][2 * c2] = w;
        }
        __syncthreads();

        v16bf a = ldfragA(&Wl[wave * 16 + ln][0], 0, h);
        #pragma unroll
        for (int j = 0; j < 4; ++j) {
            v16bf bm = ldfragB(&Xl[j * 16 + ln][0], 0, h);
            acc[j] = __builtin_amdgcn_wmma_f32_16x16x32_bf16(
                false, a, false, bm, (short)0, acc[j], false, false);
        }
        __syncthreads();
    }

    #pragma unroll
    for (int j = 0; j < 4; ++j) {
        #pragma unroll
        for (int r = 0; r < 8; ++r) {
            int o   = m0 + wave * 16 + 8 * h + r;
            int tok = tok0 + j * 16 + ln;
            float val = fmaxf(acc[j][r] + bo[o], 0.0f);   // ReLU
            y[((size_t)b * COUT + o) * TOK + tok] = val;
        }
    }
}

// =====================================================================
extern "C" void kernel_launch(void* const* d_in, const int* in_sizes, int n_in,
                              void* d_out, int out_size, void* d_ws, size_t ws_size,
                              hipStream_t stream) {
    const float* x   = (const float*)d_in[0];
    const float* tem = (const float*)d_in[1];
    const float* Wq  = (const float*)d_in[2];
    const float* bq  = (const float*)d_in[3];
    const float* Wk  = (const float*)d_in[4];
    const float* bk  = (const float*)d_in[5];
    const float* Wv  = (const float*)d_in[6];
    const float* bv  = (const float*)d_in[7];
    const float* Wo  = (const float*)d_in[8];
    const float* bo  = (const float*)d_in[9];
    float* y = (float*)d_out;

    unsigned short* ws  = (unsigned short*)d_ws;
    const size_t QSZ    = (size_t)BATCH * COUT * TOK;  // elements per tensor
    unsigned short* qkv = ws;                          // 3 * QSZ bf16
    unsigned short* ctx = ws + 3 * QSZ;                // 1 * QSZ bf16

    dim3 g1(NN, HEADS, BATCH);                         // (170, 8, 16) fused QKV
    k_qkv<<<g1, 128, 0, stream>>>(x, tem, Wq, bq, Wk, bk, Wv, bv, qkv);

    dim3 g2(NN, BATCH, HEADS);                         // (170, 16, 8)
    k_attn<<<g2, 256, 0, stream>>>(qkv, ctx);

    dim3 g3(NN, COUT / 64, BATCH);                     // (170, 8, 16)
    k_proj<<<g3, 128, 0, stream>>>(ctx, Wo, bo, y);
}
